// LifelineGNN_19911468384345
// MI455X (gfx1250) — matrix operand order repaired and checked by reading, not compiled, not run on hardware
//
#include <hip/hip_runtime.h>
#include <hip/hip_bf16.h>
#include <math.h>

typedef _Float16 f16;
typedef __attribute__((ext_vector_type(16))) _Float16 v16h;
typedef __attribute__((ext_vector_type(8)))  float    v8f;

#define Bb   2
#define Nn   10000
#define Tt   12
#define CINc 2
#define Ee   160000
#define Dd   64
#define Hh   4
#define TFo  14
#define NG   (Bb*Tt)      // 24 replicated graphs
#define ET   (Ee+Nn)      // edges incl self loops = 170000
#define RR   ((long)NG*Nn)  // 240000 GAT rows
#define RG   (Bb*Nn)        // 20000 GRU rows

#define GRU_WAVES 5                 // 5 waves/block -> 250 blocks x 5 = 1250 tiles
#define WTS (3*64*64)               // 12288 f16 per GRU weight matrix
#define SMEM_ELEMS (4*WTS + GRU_WAVES*2*1024)   // weights + per-wave h0/h1 state

// ---------- helpers ----------
__device__ __forceinline__ unsigned f2ord(float x){
  unsigned u = __float_as_uint(x);
  return (u & 0x80000000u) ? ~u : (u | 0x80000000u);
}
__device__ __forceinline__ float ord2f(unsigned u){
  return (u & 0x80000000u) ? __uint_as_float(u ^ 0x80000000u) : __uint_as_float(~u);
}
__device__ __forceinline__ v8f wmma_f16(v16h a, v16h b, v8f c){
  return __builtin_amdgcn_wmma_f32_16x16x32_f16(false, a, false, b, (short)0, c, false, false);
}
// B fragment (32x16 f16, KxN): lane col n=base+l%16, K runs [kcol..kcol+7] and [kcol+16..kcol+23],
// kcol includes +8 for lanes>=16. w is row-major [n][64] (works for global or LDS).
__device__ __forceinline__ v16h loadB(const f16* w, int nrow, int kcol){
  const f16* p = w + nrow*64 + kcol;
  v16h b;
#pragma unroll
  for (int i=0;i<8;i++){ b[i] = p[i]; b[i+8] = p[16+i]; }
  return b;
}

// ---------- tiny prep kernels ----------
__global__ void prep_scalars(const float* ew, const float* We0, const float* ae0,
                             const float* We1, const float* ae1, float* sc){
  __shared__ float red[256];
  int tid = threadIdx.x;
  float s = 0.f;
  for (int i=tid;i<Ee;i+=256) s += ew[i];
  red[tid]=s; __syncthreads();
  for (int o=128;o>0;o>>=1){ if(tid<o) red[tid]+=red[tid+o]; __syncthreads(); }
  if (tid==0) sc[0] = red[0]/(float)Ee;
  if (tid<8){
    int h = tid & 3;
    const float* W  = (tid<4)? We0 : We1;
    const float* ae = (tid<4)? ae0 : ae1;
    float c=0.f;
    for (int k=0;k<16;k++) c += W[h*16+k]*ae[h*16+k];
    sc[1+tid]=c;
  }
}

__global__ void prep_weights(const float* wih0,const float* whh0,const float* wih1,
                             const float* whh1,const float* g1W,
                             f16* o0,f16* o1,f16* o2,f16* o3,f16* og){
  int i = blockIdx.x*blockDim.x + threadIdx.x;
  if (i < WTS){ o0[i]=(f16)wih0[i]; o1[i]=(f16)whh0[i]; o2[i]=(f16)wih1[i]; o3[i]=(f16)whh1[i]; }
  if (i < Dd*Dd){ int n=i>>6, k=i&63; og[n*64+k]=(f16)g1W[k*64+n]; } // transpose to [n][k]
}

__global__ void init_kernel(float* out, unsigned* segmax, float* segsum){
  long i = (long)blockIdx.x*256 + threadIdx.x;
  if (i < RR*64) out[i] = 0.f;
  if (i < RR*4){ segmax[i] = 0x007FFFFFu; /* mapped(-inf) */ segsum[i] = 0.f; }
}

// ---------- GAT layer 0 features (CIN=2 -> 64) + alphas ----------
__global__ void feat0_kernel(const float* xseq, const float* W,
                             const float* as0, const float* ad0,
                             float* xp, float* asrc, float* adst){
  long idx = (long)blockIdx.x*blockDim.x + threadIdx.x;
  if (idx >= RR) return;
  int g = (int)(idx / Nn), n = (int)(idx % Nn);
  int bI = g / Tt, t = g % Tt;
  const float* xin = xseq + (((long)(bI*Nn+n))*Tt + t)*CINc;
  float x0 = xin[0], x1 = xin[1];
  float sa[4] = {0,0,0,0}, da[4] = {0,0,0,0};
  float* orow = xp + idx*64;
#pragma unroll
  for (int d=0; d<64; d++){
    float v = x0*W[d] + x1*W[64+d];
    orow[d] = v;
    sa[d>>4] += v*as0[d];
    da[d>>4] += v*ad0[d];
  }
#pragma unroll
  for (int h=0;h<4;h++){ asrc[idx*4+h]=sa[h]; adst[idx*4+h]=da[h]; }
}

// ---------- edge softmax/scatter passes ----------
__device__ __forceinline__ void edge_common(int g, int e, const int* ei, const float* ew,
    const float* asrc, const float* adst, const float* sc, int coff,
    int& src, int& dst, float al[4]){
  float ea;
  if (e < Ee){ src = ei[e]; dst = ei[Ee+e]; ea = ew[e]; }
  else { src = dst = e - Ee; ea = sc[0]; }
  const float* pa = asrc + ((long)g*Nn + src)*4;
  const float* pd = adst + ((long)g*Nn + dst)*4;
#pragma unroll
  for (int h=0;h<4;h++){
    float a = pa[h] + pd[h] + ea*sc[coff+h];
    al[h] = a > 0.f ? a : 0.2f*a;                 // leaky_relu(0.2)
  }
}

__global__ void edge_max_kernel(const int* ei,const float* ew,const float* asrc,
    const float* adst,const float* sc,int coff,unsigned* segmax){
  int e = blockIdx.x*256 + threadIdx.x; if (e >= ET) return;
  int g = blockIdx.y;
  int src,dst; float al[4];
  edge_common(g,e,ei,ew,asrc,adst,sc,coff,src,dst,al);
  unsigned* pm = segmax + ((long)g*Nn+dst)*4;
#pragma unroll
  for (int h=0;h<4;h++) atomicMax(&pm[h], f2ord(al[h]));
}

__global__ void edge_sum_kernel(const int* ei,const float* ew,const float* asrc,
    const float* adst,const float* sc,int coff,const unsigned* segmax,float* segsum){
  int e = blockIdx.x*256 + threadIdx.x; if (e >= ET) return;
  int g = blockIdx.y;
  int src,dst; float al[4];
  edge_common(g,e,ei,ew,asrc,adst,sc,coff,src,dst,al);
  long base = ((long)g*Nn+dst)*4;
#pragma unroll
  for (int h=0;h<4;h++){
    float m = ord2f(segmax[base+h]);
    atomicAdd(&segsum[base+h], expf(al[h]-m));
  }
}

__global__ void edge_scatter_kernel(const int* ei,const float* ew,const float* asrc,
    const float* adst,const float* sc,int coff,const unsigned* segmax,
    const float* segsum,const float* xp,float* out){
  int e = blockIdx.x*256 + threadIdx.x; if (e >= ET) return;
  int g = blockIdx.y;
  int src,dst; float al[4];
  edge_common(g,e,ei,ew,asrc,adst,sc,coff,src,dst,al);
  long base = ((long)g*Nn+dst)*4;
  float w[4];
#pragma unroll
  for (int h=0;h<4;h++){
    float m = ord2f(segmax[base+h]);
    w[h] = expf(al[h]-m) / (segsum[base+h] + 1e-16f);
  }
  const float* xr = xp + ((long)g*Nn+src)*64;
  float* orow = out + ((long)g*Nn+dst)*64;
  __builtin_prefetch(xr, 0, 0);     // global_prefetch_b8
#pragma unroll 8
  for (int c=0;c<64;c++) atomicAdd(&orow[c], xr[c]*w[c>>4]);
}

// ---------- LayerNorm + skip/residual + ReLU (in-place) ----------
__global__ void ln_relu_kernel(float* hbuf, const float* gatb,
    const float* lng, const float* lnb,
    const float* xseq, const float* skW, const float* skb,
    const float* resid, int mode){
  long idx = (long)blockIdx.x*blockDim.x + threadIdx.x;
  if (idx >= RR) return;
  int g = (int)(idx / Nn), n = (int)(idx % Nn);
  float* hrow = hbuf + idx*64;
  float s=0.f, s2=0.f;
#pragma unroll
  for (int d=0;d<64;d++){ float v=hrow[d]+gatb[d]; s+=v; s2+=v*v; }
  float m = s*(1.f/64.f);
  float var = s2*(1.f/64.f) - m*m;
  float inv = rsqrtf(var + 1e-5f);
  float x0=0.f, x1=0.f;
  if (mode==0){
    int bI=g/Tt, t=g%Tt;
    const float* xin = xseq + (((long)(bI*Nn+n))*Tt + t)*CINc;
    x0=xin[0]; x1=xin[1];
  }
#pragma unroll
  for (int d=0;d<64;d++){
    float v = hrow[d]+gatb[d];
    float y = (v-m)*inv*lng[d] + lnb[d];
    y += (mode==0) ? (x0*skW[d] + x1*skW[64+d] + skb[d]) : resid[idx*64+d];
    hrow[d] = y>0.f ? y : 0.f;
  }
}

// ---------- GAT layer 1 feature transform: xp1 = h0 @ g1_W, + alphas (WMMA) ----------
__global__ __launch_bounds__(32) void xp1_wmma_kernel(const float* h0f, const f16* wt,
    const float* as1, const float* ad1, float* xp1, float* asrcb, float* adstb){
  int l = threadIdx.x;
  int row = l & 15, hi = l >> 4, h8 = hi*8;
  long r0 = (long)blockIdx.x * 16;
  // A fragments (16x32 f16): lane row = l%16; K runs [k0+h8..+7] and [k0+16+h8..+7]
  v16h a[2];
  const float* ap = h0f + (r0 + row)*64;
#pragma unroll
  for (int ks=0; ks<2; ks++){
#pragma unroll
    for (int i=0;i<8;i++){
      a[ks][i]   = (f16)ap[ks*32 + h8 + i];
      a[ks][i+8] = (f16)ap[ks*32 + 16 + h8 + i];
    }
  }
  float* op = xp1 + r0*64;
#pragma unroll
  for (int jt=0;jt<4;jt++){
    v8f acc = {};
#pragma unroll
    for (int ks=0;ks<2;ks++)
      acc = wmma_f16(a[ks], loadB(wt, jt*16 + row, ks*32 + hi*8), acc);
    // store xp1 tile (C layout: vgpr v -> row v(+8), col = jt*16 + l%16)
#pragma unroll
    for (int v=0;v<8;v++)
      op[(long)(v+h8)*64 + jt*16 + row] = acc[v];
    // head jt == attention head; reduce over 16 channels with shfl_xor
    float asw = as1[jt*16 + row];
    float adw = ad1[jt*16 + row];
#pragma unroll
    for (int v=0;v<8;v++){
      float s = acc[v]*asw, d = acc[v]*adw;
#pragma unroll
      for (int msk=1; msk<16; msk<<=1){
        s += __shfl_xor(s, msk, 32);
        d += __shfl_xor(d, msk, 32);
      }
      if (row == 0){
        long rr = r0 + v + h8;
        asrcb[rr*4 + jt] = s;
        adstb[rr*4 + jt] = d;
      }
    }
  }
}

// ---------- fused 2-layer GRU + head (WMMA, weights staged in LDS) ----------
__device__ __forceinline__ void gru_step(const v16h ax[2], const v16h ah[2],
    const f16* wih, const f16* whh, const float* bih, const float* bhh,
    float hr[4][8], f16* hs, int row, int hi){
  int h8 = hi*8;
#pragma unroll
  for (int j=0;j<4;j++){
    int col = j*16 + row;
    v8f gi0={},gi1={},gi2={},gh0={},gh1={},gh2={};
#pragma unroll
    for (int ks=0;ks<2;ks++){
      int kcol = ks*32 + hi*8;
      gi0 = wmma_f16(ax[ks], loadB(wih, 0*64+col, kcol), gi0);
      gi1 = wmma_f16(ax[ks], loadB(wih, 1*64+col, kcol), gi1);
      gi2 = wmma_f16(ax[ks], loadB(wih, 2*64+col, kcol), gi2);
      gh0 = wmma_f16(ah[ks], loadB(whh, 0*64+col, kcol), gh0);
      gh1 = wmma_f16(ah[ks], loadB(whh, 1*64+col, kcol), gh1);
      gh2 = wmma_f16(ah[ks], loadB(whh, 2*64+col, kcol), gh2);
    }
    float bir=bih[col], biz=bih[64+col], bin=bih[128+col];
    float bhr=bhh[col], bhz=bhh[64+col], bhn=bhh[128+col];
#pragma unroll
    for (int v=0;v<8;v++){
      float r    = 1.f/(1.f + expf(-(gi0[v]+bir + gh0[v]+bhr)));
      float z    = 1.f/(1.f + expf(-(gi1[v]+biz + gh1[v]+bhz)));
      float cand = tanhf(gi2[v]+bin + r*(gh2[v]+bhn));
      float hn   = (1.f - z)*cand + z*hr[j][v];
      hr[j][v] = hn;
      hs[(v+h8)*64 + col] = (f16)hn;
    }
  }
}

__global__ __launch_bounds__(GRU_WAVES*32) void gru_head_kernel(const float* h1f,
    const f16* wih0,const f16* whh0,const f16* wih1,const f16* whh1,
    const float* bih0,const float* bhh0,const float* bih1,const float* bhh1,
    const float* headW,const float* headb,float* out){
  extern __shared__ f16 smem[];
  f16* wih0s = smem;
  f16* whh0s = smem + 1*WTS;
  f16* wih1s = smem + 2*WTS;
  f16* whh1s = smem + 3*WTS;

  int l = threadIdx.x & 31;
  int w = threadIdx.x >> 5;
  f16* h0s = smem + 4*WTS + w*2048;
  f16* h1s = h0s + 1024;

  // stage the 4 GRU weight matrices (96KB f16) into LDS once per block
  {
    unsigned* d0 = (unsigned*)wih0s; const unsigned* s0 = (const unsigned*)wih0;
    unsigned* d1 = (unsigned*)whh0s; const unsigned* s1 = (const unsigned*)whh0;
    unsigned* d2 = (unsigned*)wih1s; const unsigned* s2 = (const unsigned*)wih1;
    unsigned* d3 = (unsigned*)whh1s; const unsigned* s3 = (const unsigned*)whh1;
    for (int i = threadIdx.x; i < WTS/2; i += GRU_WAVES*32){
      d0[i]=s0[i]; d1[i]=s1[i]; d2[i]=s2[i]; d3[i]=s3[i];
    }
  }
  // zero per-wave hidden state
  for (int i=l;i<1024;i+=32){ h0s[i]=(f16)0.f; h1s[i]=(f16)0.f; }
  __syncthreads();

  int row = l & 15, hi = l >> 4, h8 = hi*8;
  long tile = (long)blockIdx.x*GRU_WAVES + w;     // exactly 1250 tiles, no idle waves
  long r0 = tile * 16;
  int b  = (int)(r0 / Nn);
  int nb = (int)(r0 % Nn);
  float h0r[4][8], h1r[4][8];
#pragma unroll
  for (int j=0;j<4;j++)
#pragma unroll
    for (int v=0;v<8;v++){ h0r[j][v]=0.f; h1r[j][v]=0.f; }

  for (int t=0;t<Tt;t++){
    // GRU0: A_x from global f32 sequence, A_h from h0 LDS
    const float* xp_ = h1f + (((long)(b*Tt+t)*Nn) + nb + row)*64;
    v16h ax[2], ah[2];
#pragma unroll
    for (int ks=0;ks<2;ks++){
#pragma unroll
      for (int i=0;i<8;i++){
        ax[ks][i]   = (f16)xp_[ks*32 + h8 + i];
        ax[ks][i+8] = (f16)xp_[ks*32 + 16 + h8 + i];
        ah[ks][i]   = h0s[row*64 + ks*32 + h8 + i];
        ah[ks][i+8] = h0s[row*64 + ks*32 + 16 + h8 + i];
      }
    }
    __syncthreads();
    gru_step(ax, ah, wih0s, whh0s, bih0, bhh0, h0r, h0s, row, hi);
    __syncthreads();
    // GRU1: A_x from fresh h0 LDS, A_h from h1 LDS
    v16h ax1[2], ah1[2];
#pragma unroll
    for (int ks=0;ks<2;ks++){
#pragma unroll
      for (int i=0;i<8;i++){
        ax1[ks][i]   = h0s[row*64 + ks*32 + h8 + i];
        ax1[ks][i+8] = h0s[row*64 + ks*32 + 16 + h8 + i];
        ah1[ks][i]   = h1s[row*64 + ks*32 + h8 + i];
        ah1[ks][i+8] = h1s[row*64 + ks*32 + 16 + h8 + i];
      }
    }
    __syncthreads();
    gru_step(ax1, ah1, wih1s, whh1s, bih1, bhh1, h1r, h1s, row, hi);
    __syncthreads();
  }
  // head: out = hT @ head_W + head_b   (16 x 14 per tile)
  for (int idx=l; idx<16*TFo; idx+=32){
    int rr = idx / TFo, tf = idx % TFo;
    float s = headb[tf];
#pragma unroll
    for (int k=0;k<64;k++) s += (float)h1s[rr*64+k]*headW[k*TFo+tf];
    out[(r0+rr)*TFo + tf] = s;
  }
}

// ---------- launcher ----------
extern "C" void kernel_launch(void* const* d_in, const int* in_sizes, int n_in,
                              void* d_out, int out_size, void* d_ws, size_t ws_size,
                              hipStream_t stream){
  (void)in_sizes; (void)n_in; (void)out_size; (void)ws_size;
  const float* x_seq = (const float*)d_in[0];
  const int*   ei    = (const int*)d_in[1];
  const float* ew    = (const float*)d_in[2];
  const float* g0W=(const float*)d_in[3],  *g0as=(const float*)d_in[4],  *g0ad=(const float*)d_in[5];
  const float* g0We=(const float*)d_in[6], *g0ae=(const float*)d_in[7],  *g0b=(const float*)d_in[8];
  const float* g1W=(const float*)d_in[9],  *g1as=(const float*)d_in[10], *g1ad=(const float*)d_in[11];
  const float* g1We=(const float*)d_in[12],*g1ae=(const float*)d_in[13], *g1b=(const float*)d_in[14];
  const float* ln0g=(const float*)d_in[15],*ln0b=(const float*)d_in[16];
  const float* ln1g=(const float*)d_in[17],*ln1b=(const float*)d_in[18];
  const float* skW=(const float*)d_in[19], *skb=(const float*)d_in[20];
  const float* gru0Wih=(const float*)d_in[21],*gru0Whh=(const float*)d_in[22];
  const float* gru0bih=(const float*)d_in[23],*gru0bhh=(const float*)d_in[24];
  const float* gru1Wih=(const float*)d_in[25],*gru1Whh=(const float*)d_in[26];
  const float* gru1bih=(const float*)d_in[27],*gru1bhh=(const float*)d_in[28];
  const float* headW=(const float*)d_in[29],*headb=(const float*)d_in[30];
  float* out = (float*)d_out;

  char* wsp = (char*)d_ws;
  auto take = [&](size_t bytes)->char*{
    char* p = wsp; wsp += (bytes + 255) & ~(size_t)255; return p;
  };
  float*    bufA   = (float*)take((size_t)RR*64*4);  // xp0 -> xp1
  float*    bufB   = (float*)take((size_t)RR*64*4);  // out0 -> h0_final (residual)
  float*    bufC   = (float*)take((size_t)RR*64*4);  // out1 -> h1_final (GRU input)
  float*    asrc   = (float*)take((size_t)RR*4*4);
  float*    adst   = (float*)take((size_t)RR*4*4);
  unsigned* segmax = (unsigned*)take((size_t)RR*4*4);
  float*    segsum = (float*)take((size_t)RR*4*4);
  float*    sc     = (float*)take(256);
  f16* wih0h = (f16*)take(WTS*2);
  f16* whh0h = (f16*)take(WTS*2);
  f16* wih1h = (f16*)take(WTS*2);
  f16* whh1h = (f16*)take(WTS*2);
  f16* g1wt  = (f16*)take(64*64*2);

  const int rb  = (int)((RR + 255)/256);
  dim3 eg((ET + 255)/256, NG);
  const int ib  = (int)((RR*64 + 255)/256);

  prep_scalars<<<1,256,0,stream>>>(ew, g0We, g0ae, g1We, g1ae, sc);
  prep_weights<<<(WTS+255)/256,256,0,stream>>>(gru0Wih,gru0Whh,gru1Wih,gru1Whh,g1W,
                                               wih0h,whh0h,wih1h,whh1h,g1wt);
  // ---- GAT layer 0 ----
  feat0_kernel<<<rb,256,0,stream>>>(x_seq, g0W, g0as, g0ad, bufA, asrc, adst);
  init_kernel<<<ib,256,0,stream>>>(bufB, segmax, segsum);
  edge_max_kernel    <<<eg,256,0,stream>>>(ei,ew,asrc,adst,sc,1,segmax);
  edge_sum_kernel    <<<eg,256,0,stream>>>(ei,ew,asrc,adst,sc,1,segmax,segsum);
  edge_scatter_kernel<<<eg,256,0,stream>>>(ei,ew,asrc,adst,sc,1,segmax,segsum,bufA,bufB);
  ln_relu_kernel<<<rb,256,0,stream>>>(bufB, g0b, ln0g, ln0b, x_seq, skW, skb, nullptr, 0);
  // ---- GAT layer 1 ----
  xp1_wmma_kernel<<<(int)(RR/16),32,0,stream>>>(bufB, g1wt, g1as, g1ad, bufA, asrc, adst);
  init_kernel<<<ib,256,0,stream>>>(bufC, segmax, segsum);
  edge_max_kernel    <<<eg,256,0,stream>>>(ei,ew,asrc,adst,sc,5,segmax);
  edge_sum_kernel    <<<eg,256,0,stream>>>(ei,ew,asrc,adst,sc,5,segmax,segsum);
  edge_scatter_kernel<<<eg,256,0,stream>>>(ei,ew,asrc,adst,sc,5,segmax,segsum,bufA,bufC);
  ln_relu_kernel<<<rb,256,0,stream>>>(bufC, g1b, ln1g, ln1b, x_seq, skW, skb, bufB, 1);
  // ---- fused GRU x2 + head (dynamic LDS: 4 weight mats + per-wave h state) ----
  gru_head_kernel<<<RG/(16*GRU_WAVES), GRU_WAVES*32, SMEM_ELEMS*sizeof(f16), stream>>>(
      bufC, wih0h,whh0h,wih1h,whh1h,
      gru0bih,gru0bhh,gru1bih,gru1bhh,
      headW, headb, out);
}
